// MultiHeadAttention_24541443129756
// MI455X (gfx1250) — matrix operand order repaired and checked by reading, not compiled, not run on hardware
//
#include <hip/hip_runtime.h>

// ---------------------------------------------------------------------------
// MI455X (gfx1250) multi-head attention, bf16 WMMA pipeline.
//   1) Q/K/V projections: fp32 GEMM via v_wmma_f32_16x16x32_bf16 -> bf16 ws
//   2) flash attention (online softmax) via WMMA, bf16 attn -> ws
//   3) output projection: bf16 x fp32W GEMM -> fp32 d_out
// CDNA5 paths: v_wmma_f32_16x16x32_bf16, global_load_async_to_lds_b64/b128
// (ASYNCcnt), split s_wait_* counters, wave32 ds_bpermute reductions.
// ---------------------------------------------------------------------------

typedef __attribute__((ext_vector_type(16))) __bf16 v16bf;
typedef __attribute__((ext_vector_type(8)))  float  v8f;
typedef int v2i __attribute__((vector_size(8)));    // matches async builtin params
typedef int v4i __attribute__((vector_size(16)));
typedef unsigned short bfstore; // bf16 bit-storage (host-safe)

constexpr int BSZ = 4;
constexpr int SEQ = 2048;
constexpr int DIM = 1024;
constexpr int NH  = 16;
constexpr int HD  = 64;
constexpr int MTOT = BSZ * SEQ;        // 8192 rows in all GEMMs

#define AS1 __attribute__((address_space(1)))
#define AS3 __attribute__((address_space(3)))

#if defined(__HIP_DEVICE_COMPILE__) && \
    __has_builtin(__builtin_amdgcn_global_load_async_to_lds_b128) && \
    __has_builtin(__builtin_amdgcn_global_load_async_to_lds_b64)
#define USE_ASYNC_LDS 1
#else
#define USE_ASYNC_LDS 0
#endif

__device__ __forceinline__ void wait_async_all() {
#if USE_ASYNC_LDS
#if __has_builtin(__builtin_amdgcn_s_wait_asynccnt)
  __builtin_amdgcn_s_wait_asynccnt(0);
#else
  asm volatile("s_wait_asynccnt 0x0" ::: "memory");
#endif
#endif
}

union AFrag { v16bf v; uint4 q[2]; };  // 16x32 bf16 A operand (8 VGPRs)
union BFrag { v16bf v; uint4 q[2]; };  // 32x16 bf16 B operand (8 VGPRs)

__device__ __forceinline__ v8f wmma_bf16(const AFrag& a, const BFrag& b, v8f c) {
  // v_wmma_f32_16x16x32_bf16 : D = A(16x32) * B(32x16) + C
  return __builtin_amdgcn_wmma_f32_16x16x32_bf16(
      /*neg_a=*/false, a.v, /*neg_b=*/false, b.v,
      /*c_mod=*/(short)0, c, /*reuse_a=*/false, /*reuse_b=*/false);
}

__device__ __forceinline__ bfstore f2bf(float f) {
  unsigned int u = __float_as_uint(f);
  unsigned int r = (u + 0x7FFFu + ((u >> 16) & 1u)) >> 16;   // RNE
  return (bfstore)r;
}

// ---------------------------------------------------------------------------
// GEMM: C[m,n] = sum_k A[m,k] * W[n,k] + bias[n]   (torch Linear, W is [N,K])
//   M = 8192, N = 1024, K = 1024.  Workgroup tile 128x128, 8 waves (wave32),
//   each wave computes 64x32 via a 4x2 grid of 16x16 WMMA accumulators.
//   A_BF16: A is bf16 (attention output) else fp32 (x / y).
//   SPLIT_OUT: write bf16 head-split [B,H,SEQ,HD]; else fp32 flat [M,N].
// ---------------------------------------------------------------------------
template <bool A_BF16, bool SPLIT_OUT>
__global__ __launch_bounds__(256)
void gemm_bt_kernel(const void* __restrict__ Ap,
                    const float* __restrict__ W,
                    const float* __restrict__ bias,
                    void* __restrict__ Out) {
  constexpr int KD = DIM;
  constexpr int LDA = 40;                 // 32 + 8 pad (keeps 16B alignment)
  __shared__ bfstore As[128 * LDA];
  __shared__ bfstore Bs[128 * LDA];

  const int t    = threadIdx.x;
  const int wave = t >> 5;
  const int lane = t & 31;
  const int ln   = lane & 15;
  const int hi   = lane >> 4;
  const int wm   = wave >> 2;             // 0..1  (M direction, 64 rows)
  const int wn   = wave & 3;              // 0..3  (N direction, 32 cols)
  const int m0   = blockIdx.x * 128;
  const int n0   = blockIdx.y * 128;

  v8f acc[4][2];
  const v8f vzero = {0.f,0.f,0.f,0.f,0.f,0.f,0.f,0.f};
  for (int i = 0; i < 4; ++i)
    for (int j = 0; j < 2; ++j) acc[i][j] = vzero;

  for (int k0 = 0; k0 < KD; k0 += 32) {
    // ---- cooperative tile loads: 128x32 of A and of W ----
    for (int cc = 0; cc < 4; ++cc) {
      const int chunk = t + 256 * cc;     // 0..1023 chunks of 4 elements
      const int row   = chunk >> 3;       // 0..127
      const int kk    = (chunk & 7) * 4;  // 0..28
      if (A_BF16) {
        const bfstore* Ab = (const bfstore*)Ap;
        const bfstore* src = Ab + (size_t)(m0 + row) * KD + k0 + kk;
        bfstore* dst = &As[row * LDA + kk];
#if USE_ASYNC_LDS
        // memory -> LDS direct (8B), tracked with ASYNCcnt
        __builtin_amdgcn_global_load_async_to_lds_b64(
            (AS1 v2i*)src, (AS3 v2i*)dst, 0, 0);
#else
        *(uint2*)dst = *(const uint2*)src;
#endif
      } else {
        const float* Af = (const float*)Ap;
        const float4 a4 = *(const float4*)(Af + (size_t)(m0 + row) * KD + k0 + kk);
        uint2 st;
        st.x = (unsigned)f2bf(a4.x) | ((unsigned)f2bf(a4.y) << 16);
        st.y = (unsigned)f2bf(a4.z) | ((unsigned)f2bf(a4.w) << 16);
        *(uint2*)&As[row * LDA + kk] = st;
      }
      const float4 b4 = *(const float4*)(W + (size_t)(n0 + row) * KD + k0 + kk);
      uint2 st;
      st.x = (unsigned)f2bf(b4.x) | ((unsigned)f2bf(b4.y) << 16);
      st.y = (unsigned)f2bf(b4.z) | ((unsigned)f2bf(b4.w) << 16);
      *(uint2*)&Bs[row * LDA + kk] = st;
    }
    if (A_BF16) wait_async_all();         // async LDS writes visible pre-barrier
    __syncthreads();

    // ---- fragments (ISA 7.12.2 layouts) ----
    AFrag af[4];
    for (int i = 0; i < 4; ++i) {
      const bfstore* p = &As[(wm * 64 + i * 16 + ln) * LDA];
      af[i].q[0] = *(const uint4*)(p + hi * 8);        // K = hi*8   .. +7
      af[i].q[1] = *(const uint4*)(p + 16 + hi * 8);   // K = 16+hi*8.. +7
    }
    BFrag bf[2];
    for (int j = 0; j < 2; ++j) {
      const bfstore* p = &Bs[(wn * 32 + j * 16 + ln) * LDA + hi * 8];
      bf[j].q[0] = *(const uint4*)(p);                 // K = hi*8 .. +7
      bf[j].q[1] = *(const uint4*)(p + 16);            // K = 16+hi*8..+7
    }
    for (int i = 0; i < 4; ++i)
      for (int j = 0; j < 2; ++j)
        acc[i][j] = wmma_bf16(af[i], bf[j], acc[i][j]);
    __syncthreads();
  }

  // ---- epilogue: bias + store (C layout: M = hi*8+r in VGPRs, N = ln) ----
  for (int i = 0; i < 4; ++i) {
    const int mbase = m0 + wm * 64 + i * 16 + hi * 8;
    for (int j = 0; j < 2; ++j) {
      const int n  = n0 + wn * 32 + j * 16 + ln;
      const float bv = bias[n];
      for (int r = 0; r < 8; ++r) {
        const float val = acc[i][j][r] + bv;
        const int m = mbase + r;
        if (SPLIT_OUT) {
          const int b = m >> 11, l = m & (SEQ - 1);      // SEQ = 2048
          const int h = n >> 6, jj = n & (HD - 1);       // HD  = 64
          ((bfstore*)Out)[(((size_t)b * NH + h) * SEQ + l) * HD + jj] = f2bf(val);
        } else {
          ((float*)Out)[(size_t)m * DIM + n] = val;
        }
      }
    }
  }
}

// ---------------------------------------------------------------------------
// Flash attention: grid = (B*H, SEQ/128), block = 256 (8 waves).
// Wave w owns 16 query rows; streams K/V in chunks of 32 with online softmax.
// K chunk goes memory->LDS via the async pipe; V is transposed through VGPRs.
// ---------------------------------------------------------------------------
__global__ __launch_bounds__(256)
void attn_kernel(const bfstore* __restrict__ Qh, const bfstore* __restrict__ Kh,
                 const bfstore* __restrict__ Vh, const int* __restrict__ mask,
                 bfstore* __restrict__ attn) {
  constexpr int LDK = 72;   // K chunk row stride (64 + 8 pad)
  constexpr int LDV = 40;   // transposed-V row stride (32 + 8 pad)
  constexpr int LDP = 40;   // P staging row stride
  __shared__ bfstore Ks[32 * LDK];       // K chunk  [k=0..31][d=0..63]
  __shared__ bfstore Vt[64 * LDV];       // V chunk transposed [d][k]
  __shared__ bfstore Ps[8][16 * LDP];    // per-wave P tile [16][32]
  __shared__ int smask[32];

  const int bh = blockIdx.x;             // b*NH + h
  const int b  = bh >> 4;
  const int h  = bh & (NH - 1);
  const int q0 = blockIdx.y * 128;
  const int t    = threadIdx.x;
  const int wave = t >> 5;
  const int lane = t & 31;
  const int ln   = lane & 15;
  const int hi   = lane >> 4;
  const size_t headbase = (size_t)bh * SEQ * HD;

  // Q fragments for this wave's 16 rows (2 K-steps of 32 over hd=64)
  AFrag qf[2];
  {
    const bfstore* qp = Qh + headbase + (size_t)(q0 + wave * 16 + ln) * HD;
    qf[0].q[0] = *(const uint4*)(qp + hi * 8);
    qf[0].q[1] = *(const uint4*)(qp + 16 + hi * 8);
    qf[1].q[0] = *(const uint4*)(qp + 32 + hi * 8);
    qf[1].q[1] = *(const uint4*)(qp + 48 + hi * 8);
  }

  v8f o_acc[4];
  const v8f vzero = {0.f,0.f,0.f,0.f,0.f,0.f,0.f,0.f};
  for (int i = 0; i < 4; ++i) o_acc[i] = vzero;
  float m_run[8], l_run[8];
  for (int r = 0; r < 8; ++r) { m_run[r] = -1e30f; l_run[r] = 0.f; }

  for (int k0 = 0; k0 < SEQ; k0 += 32) {
    // ---- cooperative chunk load: K via async-to-LDS, V transposed ----
    {
      const int row = t >> 3;                 // 0..31 key row
      const int col = (t & 7) * 8;            // 0..56 head-dim offset
      const bfstore* ksrc = Kh + headbase + (size_t)(k0 + row) * HD + col;
      bfstore* kdst = &Ks[row * LDK + col];
#if USE_ASYNC_LDS
      // global_load_async_to_lds_b128: 16B straight into LDS (ASYNCcnt)
      __builtin_amdgcn_global_load_async_to_lds_b128(
          (AS1 v4i*)ksrc, (AS3 v4i*)kdst, 0, 0);
#else
      *(uint4*)kdst = *(const uint4*)ksrc;
#endif
      const uint4 vd = *(const uint4*)(Vh + headbase + (size_t)(k0 + row) * HD + col);
      const bfstore* ve = (const bfstore*)&vd;
      for (int i = 0; i < 8; ++i) Vt[(col + i) * LDV + row] = ve[i];
      if (t < 32) smask[t] = mask[b * SEQ + k0 + t];
      if (k0 + 32 < SEQ) {                    // global_prefetch_b8 next chunk
        __builtin_prefetch(Vh + headbase + (size_t)(k0 + 32 + row) * HD + col, 0, 0);
      }
    }
    wait_async_all();
    __syncthreads();

    // ---- S = (Q @ K^T) * scale : two 16x16 tiles over the 32 keys ----
    float sc[2][8];
    for (int nt = 0; nt < 2; ++nt) {
      const bfstore* kp = &Ks[(nt * 16 + ln) * LDK];
      BFrag kf0, kf1;                       // B[d][kcol]: d = step*32+hi*8+e, step*32+16+hi*8+e
      kf0.q[0] = *(const uint4*)(kp + hi * 8);
      kf0.q[1] = *(const uint4*)(kp + 16 + hi * 8);
      kf1.q[0] = *(const uint4*)(kp + 32 + hi * 8);
      kf1.q[1] = *(const uint4*)(kp + 48 + hi * 8);
      v8f s = vzero;
      s = wmma_bf16(qf[0], kf0, s);
      s = wmma_bf16(qf[1], kf1, s);
      const bool mz = (smask[nt * 16 + ln] == 0);
      for (int r = 0; r < 8; ++r)
        sc[nt][r] = mz ? -1e30f : s[r] * 0.125f;   // 1/sqrt(64)
    }

    // ---- online softmax (row stats across the 16-lane N group) ----
    for (int r = 0; r < 8; ++r) {
      float mx = fmaxf(sc[0][r], sc[1][r]);
      for (int off = 8; off >= 1; off >>= 1)
        mx = fmaxf(mx, __shfl_xor(mx, off, 32));
      const float m_new = fmaxf(m_run[r], mx);
      const float rescale = __expf(m_run[r] - m_new);
      const float p0 = __expf(sc[0][r] - m_new);
      const float p1 = __expf(sc[1][r] - m_new);
      float rs = p0 + p1;
      for (int off = 8; off >= 1; off >>= 1)
        rs += __shfl_xor(rs, off, 32);
      l_run[r] = l_run[r] * rescale + rs;
      m_run[r] = m_new;
      for (int i = 0; i < 4; ++i) o_acc[i][r] *= rescale;
      sc[0][r] = p0; sc[1][r] = p1;
    }

    // ---- stage P through wave-private LDS: C layout -> A layout ----
    bfstore* pw = &Ps[wave][0];
    for (int nt = 0; nt < 2; ++nt)
      for (int r = 0; r < 8; ++r)
        pw[(hi * 8 + r) * LDP + nt * 16 + ln] = f2bf(sc[nt][r]);
    asm volatile("s_wait_dscnt 0x0" ::: "memory");  // wave-local LDS RAW fence

    AFrag pf;
    {
      const bfstore* pp = &Ps[wave][ln * LDP];
      pf.q[0] = *(const uint4*)(pp + hi * 8);
      pf.q[1] = *(const uint4*)(pp + 16 + hi * 8);
    }
    // ---- O += P(16x32) @ V(32x64) : 4 hd tiles ----
    for (int nt = 0; nt < 4; ++nt) {
      const bfstore* vp = &Vt[(nt * 16 + ln) * LDV + hi * 8];
      BFrag vf;
      vf.q[0] = *(const uint4*)(vp);
      vf.q[1] = *(const uint4*)(vp + 16);
      o_acc[nt] = wmma_bf16(pf, vf, o_acc[nt]);
    }
    __syncthreads();
  }

  // ---- normalize and write recombined heads: attn[b,q,h*64+j] (bf16) ----
  for (int r = 0; r < 8; ++r) {
    const float inv = 1.0f / l_run[r];
    const int q = q0 + wave * 16 + hi * 8 + r;
    for (int nt = 0; nt < 4; ++nt) {
      const int j = nt * 16 + ln;
      attn[((size_t)b * SEQ + q) * DIM + h * HD + j] = f2bf(o_acc[nt][r] * inv);
    }
  }
}

// ---------------------------------------------------------------------------
extern "C" void kernel_launch(void* const* d_in, const int* in_sizes, int n_in,
                              void* d_out, int out_size, void* d_ws, size_t ws_size,
                              hipStream_t stream) {
  (void)in_sizes; (void)n_in; (void)out_size; (void)ws_size;
  const float* x    = (const float*)d_in[0];
  const float* y    = (const float*)d_in[1];
  const int*   mask = (const int*)d_in[2];
  const float* Wq = (const float*)d_in[3];
  const float* bq = (const float*)d_in[4];
  const float* Wk = (const float*)d_in[5];
  const float* bk = (const float*)d_in[6];
  const float* Wv = (const float*)d_in[7];
  const float* bv = (const float*)d_in[8];
  const float* Wo = (const float*)d_in[9];
  const float* bo = (const float*)d_in[10];
  float* out = (float*)d_out;

  const size_t HE = (size_t)BSZ * NH * SEQ * HD;   // 8,388,608 bf16 elements
  bfstore* Qh = (bfstore*)d_ws;                    // 16 MB each, 64 MB total
  bfstore* Kh = Qh + HE;
  bfstore* Vh = Kh + HE;
  bfstore* At = Vh + HE;

  const dim3 gg(MTOT / 128, DIM / 128);            // 64 x 8 workgroups
  gemm_bt_kernel<false, true><<<gg, 256, 0, stream>>>(x, Wq, bq, Qh);
  gemm_bt_kernel<false, true><<<gg, 256, 0, stream>>>(y, Wk, bk, Kh);
  gemm_bt_kernel<false, true><<<gg, 256, 0, stream>>>(y, Wv, bv, Vh);

  attn_kernel<<<dim3(BSZ * NH, SEQ / 128), 256, 0, stream>>>(Qh, Kh, Vh, mask, At);

  gemm_bt_kernel<true, false><<<gg, 256, 0, stream>>>(At, Wo, bo, out);
}
